// HybridSpatialAttention_8907762172551
// MI455X (gfx1250) — matrix operand order repaired and checked by reading, not compiled, not run on hardware
//
#include <hip/hip_runtime.h>

// ---------------------------------------------------------------------------
// MI455X (gfx1250, wave32) implementation of HybridSpatialAttention.
// All heavy GEMMs go through v_wmma_f32_16x16x32_bf16 (f32 accumulate).
// LDS tiles are laid out so every WMMA fragment read is a contiguous
// 32-byte run per lane (vectorizes to ds_load_b128).
// ---------------------------------------------------------------------------

typedef __attribute__((ext_vector_type(16))) __bf16 bf16x16;
typedef __attribute__((ext_vector_type(8)))  __bf16 bf16x8v;
typedef __attribute__((ext_vector_type(4)))  __bf16 bf16x4v;
typedef __attribute__((ext_vector_type(8)))  float  f32x8;
typedef __attribute__((ext_vector_type(4)))  float  f32x4;

#define B_   4
#define T_   12
#define N_   512
#define D_   512
#define KH_  8
#define dh_  64
#define MTOT (B_ * T_ * N_)   // 24576 rows
#define WIN_ 12
#define DFT_ 256
#define P_   16

// WMMA fragment index maps used by this code (16-bit; contiguous k order, same for A and B):
// A (16x32): lane l -> row l%16; half = l/16;
//   element e -> K = half*16 + e                 (one contiguous 16-run)
__device__ __forceinline__ int frag_ka(int e, int half) {
  return (half << 4) + e;
}
// B (32x16): lane l -> col l%16; half = l/16;
//   element e -> K = half*16 + e                 (one contiguous 16-run)
__device__ __forceinline__ int frag_kb(int e, int half) {
  return (half << 4) + e;
}
// C/D (16x16 f32): lane l -> col l%16; VGPR j -> row = (l/16)*8 + j

__device__ __forceinline__ f32x8 wmma_bf16(bf16x16 a, bf16x16 b, f32x8 c) {
  return __builtin_amdgcn_wmma_f32_16x16x32_bf16(
      /*neg_a=*/false, a, /*neg_b=*/false, b,
      /*c_mod=*/(short)0, c, /*reuse_a=*/false, /*reuse_b=*/false);
}

__device__ __forceinline__ f32x8 zero8() {
  f32x8 z;
#pragma unroll
  for (int i = 0; i < 8; ++i) z[i] = 0.f;
  return z;
}

// ---------------------------------------------------------------------------
// Kernel 1: fused QKV projection.  out = relu([X|STE] @ W + b) -> bf16
// grid = (N/64, M/128, 3), block = 256 (8 waves). K-dim = 1024.
// As: [row][k] (128x32), Bs: [col][k] (64x32, transposed at staging).
// ---------------------------------------------------------------------------
__global__ __launch_bounds__(256)
void qkv_gemm_kernel(const float* __restrict__ X, const float* __restrict__ STE,
                     const float* __restrict__ Wq, const float* __restrict__ bq,
                     const float* __restrict__ Wk, const float* __restrict__ bk,
                     const float* __restrict__ Wv, const float* __restrict__ bv,
                     __bf16* __restrict__ Qb, __bf16* __restrict__ Kb,
                     __bf16* __restrict__ Vb) {
  const float* W;  const float* bias;  __bf16* out;
  if (blockIdx.z == 0)      { W = Wq; bias = bq; out = Qb; }
  else if (blockIdx.z == 1) { W = Wk; bias = bk; out = Kb; }
  else                      { W = Wv; bias = bv; out = Vb; }

  __shared__ __bf16 As[128 * 32];
  __shared__ __bf16 Bs[64 * 32];

  const int tid  = threadIdx.x;
  const int wave = tid >> 5, lane = tid & 31;
  const int half = lane >> 4, lr = lane & 15;
  const int m0 = blockIdx.y * 128;
  const int n0 = blockIdx.x * 64;

  f32x8 acc[4];
#pragma unroll
  for (int nt = 0; nt < 4; ++nt) acc[nt] = zero8();

  for (int kt = 0; kt < 2 * D_; kt += 32) {
    // Stage A tile (128x32) as bf16: concat([X, STE]) along K. kt is uniform,
    // and a 32-wide K slice never straddles the X/STE boundary.
    const float* Asrc = (kt < D_) ? X : STE;
    const int    koff = (kt < D_) ? kt : kt - D_;
#pragma unroll
    for (int i = 0; i < 4; ++i) {
      int f = i * 256 + tid;            // 1024 float4 = 128 rows x 8
      int r = f >> 3, c4 = f & 7;
      f32x4 v = *(const f32x4*)(Asrc + (size_t)(m0 + r) * D_ + koff + c4 * 4);
      bf16x4v bv4;
#pragma unroll
      for (int u = 0; u < 4; ++u) bv4[u] = (__bf16)v[u];
      *(bf16x4v*)&As[r * 32 + c4 * 4] = bv4;
    }
    // Stage B tile transposed: Bs[col][k].  W is [2D, D] row-major.
#pragma unroll
    for (int i = 0; i < 2; ++i) {
      int f = i * 256 + tid;            // 512 float4 = 32 rows x 16
      int r = f >> 4, c4 = f & 15;
      f32x4 v = *(const f32x4*)(W + (size_t)(kt + r) * D_ + n0 + c4 * 4);
#pragma unroll
      for (int u = 0; u < 4; ++u) Bs[(c4 * 4 + u) * 32 + r] = (__bf16)v[u];
    }
    __syncthreads();

    bf16x16 a;
#pragma unroll
    for (int e = 0; e < 16; ++e)
      a[e] = As[(wave * 16 + lr) * 32 + frag_ka(e, half)];
#pragma unroll
    for (int nt = 0; nt < 4; ++nt) {
      bf16x16 bf;
#pragma unroll
      for (int e = 0; e < 16; ++e)
        bf[e] = Bs[(nt * 16 + lr) * 32 + frag_kb(e, half)];  // contiguous 16
      acc[nt] = wmma_bf16(a, bf, acc[nt]);
    }
    __syncthreads();
  }

#pragma unroll
  for (int nt = 0; nt < 4; ++nt) {
#pragma unroll
    for (int j = 0; j < 8; ++j) {
      int row = m0 + wave * 16 + half * 8 + j;
      int col = n0 + nt * 16 + lr;
      float v = acc[nt][j] + bias[col];
      v = v > 0.f ? v : 0.f;
      out[(size_t)row * D_ + col] = (__bf16)v;
    }
  }
}

// ---------------------------------------------------------------------------
// Kernel 2: scalar[b,t,n] = X[b,t,n,:] @ W_scalar.  Wave per row.
// ---------------------------------------------------------------------------
__global__ __launch_bounds__(256)
void scalar_kernel(const float* __restrict__ X, const float* __restrict__ Wsc,
                   float* __restrict__ sc) {
  const int wave = threadIdx.x >> 5, lane = threadIdx.x & 31;
  const int m = blockIdx.x * 8 + wave;
  float p = 0.f;
#pragma unroll
  for (int i = 0; i < 16; ++i) {
    int c = lane + 32 * i;
    p += X[(size_t)m * D_ + c] * Wsc[c];
  }
#pragma unroll
  for (int off = 16; off; off >>= 1) p += __shfl_xor(p, off, 32);
  if (lane == 0) sc[m] = p;
}

// ---------------------------------------------------------------------------
// Kernel 3: mem = pats @ W_pm, val = pats @ W_pv.  grid (16, 2), block 256.
// ---------------------------------------------------------------------------
__global__ __launch_bounds__(256)
void pattern_kernel(const float* __restrict__ pk, const float* __restrict__ Wpm,
                    const float* __restrict__ Wpv, float* __restrict__ mem,
                    float* __restrict__ val) {
  const int p = blockIdx.x, j = threadIdx.x;
  const float* W = blockIdx.y ? Wpv : Wpm;
  float* o = blockIdx.y ? val : mem;
  float s = 0.f;
#pragma unroll
  for (int i = 0; i < WIN_; ++i) s += pk[p * WIN_ + i] * W[i * DFT_ + j];
  o[p * DFT_ + j] = s;
}

// ---------------------------------------------------------------------------
// Kernel 4: DFT pattern branch; adds dft_out into Q's geo heads (cols 0..255).
// Wave per row; lane owns cols {lane, lane+32, ..., lane+224}.
// ---------------------------------------------------------------------------
__global__ __launch_bounds__(256)
void dft_kernel(const float* __restrict__ sc, const float* __restrict__ Wwq,
                const float* __restrict__ mem, const float* __restrict__ val,
                __bf16* __restrict__ Qb) {
  const int wave = threadIdx.x >> 5, lane = threadIdx.x & 31;
  const int m = blockIdx.x * 8 + wave;
  const int b = m / (T_ * N_);
  const int rem = m % (T_ * N_);
  const int t = rem / N_, n = rem % N_;

  float win[WIN_];
#pragma unroll
  for (int i = 0; i < WIN_; ++i) {
    int ts = t + i - (WIN_ - 1);
    win[i] = (ts >= 0) ? sc[(size_t)(b * T_ + ts) * N_ + n] : 0.f;
  }

  float qd[8];
#pragma unroll
  for (int e = 0; e < 8; ++e) {
    int j = lane + 32 * e;
    float s = 0.f;
#pragma unroll
    for (int i = 0; i < WIN_; ++i) s += win[i] * Wwq[i * DFT_ + j];
    qd[e] = s;
  }

  float ps[P_];
#pragma unroll
  for (int p = 0; p < P_; ++p) {
    float s = 0.f;
#pragma unroll
    for (int e = 0; e < 8; ++e) s += qd[e] * mem[p * DFT_ + lane + 32 * e];
#pragma unroll
    for (int off = 16; off; off >>= 1) s += __shfl_xor(s, off, 32);
    ps[p] = s * 0.0625f;  // 1/sqrt(256)
  }

  float mx = ps[0];
#pragma unroll
  for (int p = 1; p < P_; ++p) mx = fmaxf(mx, ps[p]);
  float den = 0.f, ap[P_];
#pragma unroll
  for (int p = 0; p < P_; ++p) { ap[p] = __expf(ps[p] - mx); den += ap[p]; }
  const float inv = 1.f / den;

#pragma unroll
  for (int e = 0; e < 8; ++e) {
    int j = lane + 32 * e;
    float o = 0.f;
#pragma unroll
    for (int p = 0; p < P_; ++p) o += ap[p] * val[p * DFT_ + j];
    size_t qi = (size_t)m * D_ + j;
    Qb[qi] = (__bf16)((float)Qb[qi] + o * inv);
  }
}

// ---------------------------------------------------------------------------
// Kernel 5: masked attention, flash-style online softmax, all WMMA.
// grid = (N/128, B*KH*T), block = 256 (8 waves; wave owns 16 query rows).
// Per 32-key step: K tile (row-major) and V tile (transposed) are staged
// cooperatively in LDS (shared by all 8 waves -> 8x less global traffic,
// contiguous fragment reads); 4 WMMAs for S, masked online softmax with
// 16-lane shuffles, P re-staged through per-wave LDS, 4 WMMAs for P@V.
// ---------------------------------------------------------------------------
__global__ __launch_bounds__(256)
void attn_kernel(const __bf16* __restrict__ Qb, const __bf16* __restrict__ Kb,
                 const __bf16* __restrict__ Vb,
                 const unsigned char* __restrict__ geo,
                 const unsigned char* __restrict__ sem,
                 __bf16* __restrict__ Mg) {
  __shared__ __bf16 Ks[32 * 64];      // [key][dim]
  __shared__ __bf16 Vs[64 * 32];      // [dim][key]  (transposed at staging)
  __shared__ __bf16 Ps[8][16 * 32];   // per-wave P staging, [row][key]

  const int tid  = threadIdx.x;
  const int wave = tid >> 5, lane = tid & 31;
  const int half = lane >> 4, lr = lane & 15;
  const int y = blockIdx.y;
  const int t = y % T_;
  const int h = (y / T_) % KH_;
  const int b = y / (T_ * KH_);
  const unsigned char* mask = (h < 4) ? geo : sem;
  const size_t rowbase = (size_t)(b * T_ + t) * N_;
  const int q0 = blockIdx.x * 128 + wave * 16;
  const float scale = 0.125f;  // 1/sqrt(64)

  // Q fragments (A layout), dims 0..31 and 32..63 of this head.
  bf16x16 a0, a1;
  {
    const __bf16* qp = Qb + (rowbase + q0 + lr) * D_ + h * dh_;
#pragma unroll
    for (int e = 0; e < 16; ++e) {
      a0[e] = qp[frag_ka(e, half)];
      a1[e] = qp[32 + frag_ka(e, half)];
    }
  }

  f32x8 o[4];
#pragma unroll
  for (int nt = 0; nt < 4; ++nt) o[nt] = zero8();
  float mrow[8], lsum[8];
#pragma unroll
  for (int j = 0; j < 8; ++j) { mrow[j] = -1e30f; lsum[j] = 0.f; }

  for (int kb = 0; kb < N_; kb += 32) {
    __syncthreads();  // previous iteration's fragment reads done
    // --- stage K (row-major) and V (transposed) tiles: 32 keys x 64 dims ---
    {
      int key = tid >> 3, v8 = tid & 7;  // 256 threads = 32 keys x 8 vec8
      const size_t src = (rowbase + kb + key) * D_ + h * dh_ + v8 * 8;
      bf16x8v kv = *(const bf16x8v*)(Kb + src);
      *(bf16x8v*)&Ks[key * 64 + v8 * 8] = kv;
      bf16x8v vv = *(const bf16x8v*)(Vb + src);
#pragma unroll
      for (int u = 0; u < 8; ++u) Vs[(v8 * 8 + u) * 32 + key] = vv[u];
    }
    __syncthreads();

    // --- S = Q @ K^T for 32 keys (two 16-key C tiles) ---
    f32x8 s01[2];
#pragma unroll
    for (int s = 0; s < 2; ++s) {
      bf16x16 b0, b1;
#pragma unroll
      for (int e = 0; e < 16; ++e) {
        b0[e] = Ks[(s * 16 + lr) * 64 + frag_kb(e, half)];       // contiguous
        b1[e] = Ks[(s * 16 + lr) * 64 + 32 + frag_kb(e, half)];  // contiguous
      }
      f32x8 c = zero8();
      c = wmma_bf16(a0, b0, c);
      c = wmma_bf16(a1, b1, c);
      s01[s] = c;
    }

    // --- masked online softmax update, per owned row j ---
#pragma unroll
    for (int j = 0; j < 8; ++j) {
      const int qrow = q0 + half * 8 + j;
      const bool mk0 = mask[(size_t)qrow * N_ + kb + lr] != 0;
      const bool mk1 = mask[(size_t)qrow * N_ + kb + 16 + lr] != 0;
      float v0 = mk0 ? s01[0][j] * scale : -1e30f;
      float v1 = mk1 ? s01[1][j] * scale : -1e30f;
      float tm = fmaxf(v0, v1);
#pragma unroll
      for (int off = 8; off; off >>= 1) tm = fmaxf(tm, __shfl_xor(tm, off, 16));
      const float mnew = fmaxf(mrow[j], tm);
      const float alpha = __expf(mrow[j] - mnew);
      float p0 = mk0 ? __expf(v0 - mnew) : 0.f;  // fully-masked rows -> 0
      float p1 = mk1 ? __expf(v1 - mnew) : 0.f;
      float psum = p0 + p1;
#pragma unroll
      for (int off = 8; off; off >>= 1) psum += __shfl_xor(psum, off, 16);
      lsum[j] = lsum[j] * alpha + psum;
      mrow[j] = mnew;
#pragma unroll
      for (int nt = 0; nt < 4; ++nt) o[nt][j] *= alpha;
      const int prow = half * 8 + j;
      Ps[wave][prow * 32 + lr]      = (__bf16)p0;   // C layout -> LDS
      Ps[wave][prow * 32 + 16 + lr] = (__bf16)p1;
    }

    // --- P (A layout) from LDS; wave-private region, in-order DS ops ---
    bf16x16 pa;
#pragma unroll
    for (int e = 0; e < 16; ++e)
      pa[e] = Ps[wave][lr * 32 + frag_ka(e, half)];

    // --- O += P @ V ---
#pragma unroll
    for (int nt = 0; nt < 4; ++nt) {
      bf16x16 bv;
#pragma unroll
      for (int e = 0; e < 16; ++e)
        bv[e] = Vs[(nt * 16 + lr) * 32 + frag_kb(e, half)];      // contiguous
      o[nt] = wmma_bf16(pa, bv, o[nt]);
    }
  }

  // epilogue: normalize (0 for fully-masked rows) and store merged bf16.
#pragma unroll
  for (int j = 0; j < 8; ++j) {
    const float inv = lsum[j] > 0.f ? 1.f / lsum[j] : 0.f;
    const int row = q0 + half * 8 + j;
#pragma unroll
    for (int nt = 0; nt < 4; ++nt)
      Mg[(rowbase + row) * D_ + h * dh_ + nt * 16 + lr] =
          (__bf16)(o[nt][j] * inv);
  }
}

// ---------------------------------------------------------------------------
// Kernel 6: out = relu(merged @ Wo + bo) -> f32.  K-dim = 512.
// ---------------------------------------------------------------------------
__global__ __launch_bounds__(256)
void out_gemm_kernel(const __bf16* __restrict__ Mg, const float* __restrict__ Wo,
                     const float* __restrict__ bo, float* __restrict__ out) {
  __shared__ __bf16 As[128 * 32];
  __shared__ __bf16 Bs[64 * 32];

  const int tid  = threadIdx.x;
  const int wave = tid >> 5, lane = tid & 31;
  const int half = lane >> 4, lr = lane & 15;
  const int m0 = blockIdx.y * 128;
  const int n0 = blockIdx.x * 64;

  f32x8 acc[4];
#pragma unroll
  for (int nt = 0; nt < 4; ++nt) acc[nt] = zero8();

  for (int kt = 0; kt < D_; kt += 32) {
    // A tile: straight bf16 copy, vectorized 8-wide.
#pragma unroll
    for (int i = 0; i < 2; ++i) {
      int f = i * 256 + tid;            // 512 vec8 = 128 rows x 4
      int r = f >> 2, v8 = f & 3;
      *(bf16x8v*)&As[r * 32 + v8 * 8] =
          *(const bf16x8v*)(Mg + (size_t)(m0 + r) * D_ + kt + v8 * 8);
    }
    // B tile transposed: Bs[col][k].
#pragma unroll
    for (int i = 0; i < 2; ++i) {
      int f = i * 256 + tid;
      int r = f >> 4, c4 = f & 15;
      f32x4 v = *(const f32x4*)(Wo + (size_t)(kt + r) * D_ + n0 + c4 * 4);
#pragma unroll
      for (int u = 0; u < 4; ++u) Bs[(c4 * 4 + u) * 32 + r] = (__bf16)v[u];
    }
    __syncthreads();

    bf16x16 a;
#pragma unroll
    for (int e = 0; e < 16; ++e)
      a[e] = As[(wave * 16 + lr) * 32 + frag_ka(e, half)];
#pragma unroll
    for (int nt = 0; nt < 4; ++nt) {
      bf16x16 bf;
#pragma unroll
      for (int e = 0; e < 16; ++e)
        bf[e] = Bs[(nt * 16 + lr) * 32 + frag_kb(e, half)];
      acc[nt] = wmma_bf16(a, bf, acc[nt]);
    }
    __syncthreads();
  }

#pragma unroll
  for (int nt = 0; nt < 4; ++nt) {
#pragma unroll
    for (int j = 0; j < 8; ++j) {
      int row = m0 + wave * 16 + half * 8 + j;
      int col = n0 + nt * 16 + lr;
      float v = acc[nt][j] + bo[col];
      out[(size_t)row * D_ + col] = v > 0.f ? v : 0.f;
    }
  }
}

// ---------------------------------------------------------------------------
// Host launcher
// ---------------------------------------------------------------------------
extern "C" void kernel_launch(void* const* d_in, const int* in_sizes, int n_in,
                              void* d_out, int out_size, void* d_ws,
                              size_t ws_size, hipStream_t stream) {
  (void)in_sizes; (void)n_in; (void)out_size; (void)ws_size;

  const float* X   = (const float*)d_in[0];
  const float* STE = (const float*)d_in[1];
  const unsigned char* geo = (const unsigned char*)d_in[2];
  const unsigned char* sem = (const unsigned char*)d_in[3];
  const float* pk  = (const float*)d_in[4];
  const float* Wq  = (const float*)d_in[5];
  const float* bq  = (const float*)d_in[6];
  const float* Wk  = (const float*)d_in[7];
  const float* bk  = (const float*)d_in[8];
  const float* Wv  = (const float*)d_in[9];
  const float* bv  = (const float*)d_in[10];
  const float* Wo  = (const float*)d_in[11];
  const float* bo  = (const float*)d_in[12];
  const float* Wsc = (const float*)d_in[13];
  const float* Wwq = (const float*)d_in[14];
  const float* Wpm = (const float*)d_in[15];
  const float* Wpv = (const float*)d_in[16];

  const size_t MD = (size_t)MTOT * D_;
  __bf16* Qb = (__bf16*)d_ws;
  __bf16* Kb = Qb + MD;
  __bf16* Vb = Kb + MD;
  __bf16* Mg = Vb + MD;
  float* scal = (float*)(Mg + MD);
  float* mem  = scal + MTOT;
  float* val  = mem + P_ * DFT_;

  qkv_gemm_kernel<<<dim3(D_ / 64, MTOT / 128, 3), 256, 0, stream>>>(
      X, STE, Wq, bq, Wk, bk, Wv, bv, Qb, Kb, Vb);
  scalar_kernel<<<dim3(MTOT / 8), 256, 0, stream>>>(X, Wsc, scal);
  pattern_kernel<<<dim3(P_, 2), 256, 0, stream>>>(pk, Wpm, Wpv, mem, val);
  dft_kernel<<<dim3(MTOT / 8), 256, 0, stream>>>(scal, Wwq, mem, val, Qb);
  attn_kernel<<<dim3(N_ / 128, B_ * KH_ * T_), 256, 0, stream>>>(
      Qb, Kb, Vb, geo, sem, Mg);
  out_gemm_kernel<<<dim3(D_ / 64, MTOT / 128), 256, 0, stream>>>(
      Mg, Wo, bo, (float*)d_out);
}
